// Decoder_20040317403603
// MI455X (gfx1250) — matrix operand order repaired and checked by reading, not compiled, not run on hardware
//
#include <hip/hip_runtime.h>
#include <hip/hip_fp16.h>

typedef __attribute__((ext_vector_type(16))) _Float16 v16h;
typedef __attribute__((ext_vector_type(8)))  float    v8f;

// ---------------- problem constants ----------------
constexpr int B_ = 16, S_ = 16, W_ = 32, H_ = 512, V_ = 32000, EXTRA_ = 100, T_ = 64, BOS_ = 1;
constexpr int VEXT_ = V_ + EXTRA_;          // 32100
constexpr int SW_   = S_ * W_;              // 512
constexpr int NWROW_ = B_ * S_ * W_;        // 8192 word rows

// ---------------- workspace layout (bytes) ----------------
constexpr size_t AL(size_t x) { return (x + 255) & ~size_t(255); }
constexpr size_t OFF_OUTW16 = 0;
constexpr size_t OFF_GIH16  = AL(OFF_OUTW16 + (size_t)V_ * H_ * 2);
constexpr size_t OFF_GHH16  = AL(OFF_GIH16  + (size_t)3*H_ * 2*H_ * 2);
constexpr size_t OFF_WAW1   = AL(OFF_GHH16  + (size_t)3*H_ * H_ * 2);
constexpr size_t OFF_WAW2   = AL(OFF_WAW1   + (size_t)H_ * H_ * 2);
constexpr size_t OFF_SAW1   = AL(OFF_WAW2   + (size_t)H_ * H_ * 2);
constexpr size_t OFF_SAW2   = AL(OFF_SAW1   + (size_t)H_ * H_ * 2);
constexpr size_t OFF_SELW16 = AL(OFF_SAW2   + (size_t)H_ * H_ * 2);
constexpr size_t OFF_WORD16 = AL(OFF_SELW16 + (size_t)H_ * 3*H_ * 2);
constexpr size_t OFF_SENT16 = AL(OFF_WORD16 + (size_t)NWROW_ * H_ * 2);
constexpr size_t OFF_WAPRE  = AL(OFF_SENT16 + (size_t)B_*S_ * H_ * 2);
constexpr size_t OFF_SAPRE  = AL(OFF_WAPRE  + (size_t)NWROW_ * H_ * 4);
constexpr size_t OFF_H32    = AL(OFF_SAPRE  + (size_t)B_*S_ * H_ * 4);
constexpr size_t OFF_H16    = AL(OFF_H32    + (size_t)B_ * H_ * 4);
constexpr size_t OFF_RNN16  = AL(OFF_H16    + (size_t)B_ * H_ * 2);
constexpr size_t OFF_SEL16  = AL(OFF_RNN16  + (size_t)B_ * 2*H_ * 2);
constexpr size_t OFF_DQW    = AL(OFF_SEL16  + (size_t)B_ * 3*H_ * 2);
constexpr size_t OFF_DQS    = AL(OFF_DQW    + (size_t)B_ * H_ * 4);
constexpr size_t OFF_SCW    = AL(OFF_DQS    + (size_t)B_ * H_ * 4);
constexpr size_t OFF_SCS    = AL(OFF_SCW    + (size_t)NWROW_ * 4);
constexpr size_t OFF_RESC   = AL(OFF_SCS    + (size_t)B_*S_ * 4);
constexpr size_t OFF_GI     = AL(OFF_RESC   + (size_t)B_ * SW_ * 4);
constexpr size_t OFF_GH     = AL(OFF_GI     + (size_t)B_ * 3*H_ * 4);
constexpr size_t OFF_SELPRE = AL(OFF_GH     + (size_t)B_ * 3*H_ * 4);
constexpr size_t OFF_PGEN   = AL(OFF_SELPRE + (size_t)B_ * H_ * 4);
constexpr size_t OFF_LOGITS = AL(OFF_PGEN   + 256);

// output layout (floats)
constexpr size_t OUT_OFF_H  = (size_t)B_ * T_ * VEXT_;
constexpr size_t OUT_OFF_PG = OUT_OFF_H + (size_t)B_ * H_;

// ---------------- WMMA fragment helpers (CDNA5 16x16x32 f16 layouts) ----------------
// A (16xK row-major, f16), tile at (m0, k0). Lanes 0-15: M=lane, K 0..15;
// lanes 16-31: M=lane-16, K 16..31.
__device__ __forceinline__ v16h load_a_frag(const _Float16* __restrict__ A, int lda,
                                            int m0, int k0, int lane) {
  int m  = m0 + (lane & 15);
  int kb = (lane < 16) ? 0 : 16;
  const _Float16* row = A + (size_t)m * lda + k0;
  v16h a;
#pragma unroll
  for (int j = 0; j < 8; ++j) {
    int kk = kb + 2 * j;
    a[2 * j]     = row[kk];
    a[2 * j + 1] = row[kk + 1];
  }
  return a;
}

// B (32x16) built from row-major weight Wt (N x K): B[k][n] = Wt[n][k].
// Lane: n = n0 + (lane&15); 16 halfs = contiguous K run (kb = 0 or 16) -> one 32B load.
__device__ __forceinline__ v16h load_b_frag(const _Float16* __restrict__ Wt, int ldk,
                                            int n0, int k0, int lane) {
  int n  = n0 + (lane & 15);
  int kb = (lane < 16) ? 0 : 16;
  const _Float16* p = Wt + (size_t)n * ldk + k0 + kb;
  return *(const v16h*)p;   // ldk multiple of 16, k0 multiple of 32 -> 32B aligned
}

__device__ __forceinline__ v8f wmma_step(v16h a, v16h b, v8f c) {
  return __builtin_amdgcn_wmma_f32_16x16x32_f16(false, a, false, b, (short)0, c,
                                                false, false);
}

// ---------------- GEMM kernels ----------------
// out(16 x N) = A(16 x K) @ Wt(N x K)^T + bias ; one wave per 16x16 tile.
// Software-pipelined: fragments for k+32 are in flight while WMMA consumes k,
// so the scheduler can wait on a partial loadcnt instead of 0.
__global__ void gemm16_f16_kernel(const _Float16* __restrict__ A,
                                  const _Float16* __restrict__ Wt,
                                  const float* __restrict__ bias,
                                  float* __restrict__ out, int N, int K) {
  int wave = (blockIdx.x * blockDim.x + threadIdx.x) >> 5;
  int lane = threadIdx.x & 31;
  if (wave >= (N >> 4)) return;            // wave-uniform: EXEC stays all-ones
  int n0 = wave << 4;
  const _Float16* wrow = Wt + (size_t)(n0 + (lane & 15)) * K + ((lane < 16) ? 0 : 16);
  v8f c = {};
  v16h a0 = load_a_frag(A, K, 0, 0, lane);
  v16h b0 = load_b_frag(Wt, K, n0, 0, lane);
  for (int k0 = 0; k0 < K; k0 += 32) {
    v16h a1 = a0, b1 = b0;
    if (k0 + 32 < K) {
      a1 = load_a_frag(A, K, 0, k0 + 32, lane);
      b1 = load_b_frag(Wt, K, n0, k0 + 32, lane);
      // WGP-scope prefetch: warm L0/WGP$ from L2 two k-steps ahead
      __builtin_prefetch((const void*)(wrow + k0 + 96), 0, 3);
    }
    c = wmma_step(a0, b0, c);
    a0 = a1; b0 = b1;
  }
  int n  = n0 + (lane & 15);
  int mb = (lane < 16) ? 0 : 8;
  float bv = bias ? bias[n] : 0.0f;
#pragma unroll
  for (int j = 0; j < 8; ++j)
    out[(size_t)(mb + j) * (size_t)N + n] = c[j] + bv;
}

// out(M x N) = A(M x K) @ Wt(N x K)^T + bias ; one wave per 16x16 tile.
__global__ void gemm_tiled_f16_kernel(const _Float16* __restrict__ A,
                                      const _Float16* __restrict__ Wt,
                                      const float* __restrict__ bias,
                                      float* __restrict__ out, int M, int N, int K) {
  int wave = (blockIdx.x * blockDim.x + threadIdx.x) >> 5;
  int lane = threadIdx.x & 31;
  int ntiles = N >> 4;
  if (wave >= (M >> 4) * ntiles) return;
  int m0 = (wave / ntiles) << 4;
  int n0 = (wave % ntiles) << 4;
  v8f c = {};
  v16h a0 = load_a_frag(A, K, m0, 0, lane);
  v16h b0 = load_b_frag(Wt, K, n0, 0, lane);
  for (int k0 = 0; k0 < K; k0 += 32) {
    v16h a1 = a0, b1 = b0;
    if (k0 + 32 < K) {
      a1 = load_a_frag(A, K, m0, k0 + 32, lane);
      b1 = load_b_frag(Wt, K, n0, k0 + 32, lane);
    }
    c = wmma_step(a0, b0, c);
    a0 = a1; b0 = b1;
  }
  int n  = n0 + (lane & 15);
  int mb = (lane < 16) ? 0 : 8;
  float bv = bias ? bias[n] : 0.0f;
#pragma unroll
  for (int j = 0; j < 8; ++j)
    out[(size_t)(m0 + mb + j) * (size_t)N + n] = c[j] + bv;
}

// ---------------- reductions ----------------
__device__ __forceinline__ float wave_sum(float v) {
#pragma unroll
  for (int m = 16; m; m >>= 1) v += __shfl_xor(v, m, 32);
  return v;
}
__device__ __forceinline__ float wave_max(float v) {
#pragma unroll
  for (int m = 16; m; m >>= 1) v = fmaxf(v, __shfl_xor(v, m, 32));
  return v;
}
__device__ __forceinline__ float block_sum(float v, float* buf, int nthreads) {
  int tid = threadIdx.x;
  v = wave_sum(v);
  if ((tid & 31) == 0) buf[tid >> 5] = v;
  __syncthreads();
  if (tid < 32) {
    float r = (tid < (nthreads >> 5)) ? buf[tid] : 0.0f;
    r = wave_sum(r);
    if (tid == 0) buf[0] = r;
  }
  __syncthreads();
  float r = buf[0];
  __syncthreads();
  return r;
}
__device__ __forceinline__ float block_max(float v, float* buf, int nthreads) {
  int tid = threadIdx.x;
  v = wave_max(v);
  if ((tid & 31) == 0) buf[tid >> 5] = v;
  __syncthreads();
  if (tid < 32) {
    float r = (tid < (nthreads >> 5)) ? buf[tid] : -3.4e38f;
    r = wave_max(r);
    if (tid == 0) buf[0] = r;
  }
  __syncthreads();
  float r = buf[0];
  __syncthreads();
  return r;
}

// ---------------- elementwise / pipeline kernels ----------------
__global__ void f32_to_f16_kernel(const float* __restrict__ in,
                                  _Float16* __restrict__ out, size_t n) {
  size_t i = blockIdx.x * (size_t)blockDim.x + threadIdx.x;
  size_t st = gridDim.x * (size_t)blockDim.x;
  for (; i < n; i += st) out[i] = (_Float16)in[i];
}

__global__ void init_h_kernel(const float* __restrict__ d0, float* __restrict__ h32,
                              _Float16* __restrict__ h16) {
  int i = blockIdx.x * blockDim.x + threadIdx.x;   // B*H = 8192
  float v = d0[i];
  h32[i] = v;
  h16[i] = (_Float16)v;
}

// x = relu(emb[safe_tok]); feed into rnn_in[:, 0:H] and sel_in[:, H:2H] (f16)
__global__ void embed_kernel(const int* __restrict__ target, const float* __restrict__ emb,
                             _Float16* __restrict__ rnn16, _Float16* __restrict__ sel16,
                             int t) {
  int b = blockIdx.x, h = threadIdx.x;             // 512 threads
  int tok = (t == 0) ? BOS_ : target[b * T_ + (t - 1)];
  if (tok >= V_) tok = 2;
  if (tok < 0)  tok = 0;
  float v = emb[(size_t)tok * H_ + h];
  v = v > 0.0f ? v : 0.0f;
  _Float16 x = (_Float16)v;
  rnn16[b * (2 * H_) + h]       = x;
  sel16[b * (3 * H_) + H_ + h]  = x;
}

// out[row] = sum_h tanh(pre[row,h] + dq[b,h]) * vvec[h] + vb ; one wave per row
__global__ void attn_score_kernel(const float* __restrict__ pre, const float* __restrict__ dq,
                                  const float* __restrict__ vvec, const float* __restrict__ vb,
                                  float* __restrict__ out, int rowsPerBatch, int nRows) {
  int wave = (blockIdx.x * blockDim.x + threadIdx.x) >> 5;
  int lane = threadIdx.x & 31;
  if (wave >= nRows) return;
  int b = wave / rowsPerBatch;
  const float* prow = pre + (size_t)wave * H_;
  const float* drow = dq + (size_t)b * H_;
  float acc = 0.0f;
  for (int h = lane; h < H_; h += 32) acc += tanhf(prow[h] + drow[h]) * vvec[h];
  acc = wave_sum(acc);
  if (lane == 0) out[wave] = acc + vb[0];
}

// per-sentence softmax over W (one wave per sentence), sentence softmax over S,
// combine & renormalize -> resc (B x 512)
__global__ void attn_combine_kernel(const float* __restrict__ scw,
                                    const float* __restrict__ scs,
                                    float* __restrict__ resc) {
  __shared__ float buf[32];
  __shared__ float pas[S_];
  int b = blockIdx.x, tid = threadIdx.x;           // 512 threads = 16 waves = 16 sentences
  float v  = scw[b * SW_ + tid];
  float mx = wave_max(v);                          // softmax over W within wave
  float e  = expf(v - mx);
  float se = wave_sum(e);
  float paw = e / se;
  if (tid == 0) {                                  // sentence softmax over S=16
    float m2 = -3.4e38f;
    for (int s = 0; s < S_; ++s) m2 = fmaxf(m2, scs[b * S_ + s]);
    float s2 = 0.0f;
    for (int s = 0; s < S_; ++s) { float t2 = expf(scs[b * S_ + s] - m2); pas[s] = t2; s2 += t2; }
    for (int s = 0; s < S_; ++s) pas[s] /= s2;
  }
  __syncthreads();
  float num = paw * pas[tid >> 5];
  float tot = block_sum(num, buf, 512);
  resc[b * SW_ + tid] = num / (tot + 1e-10f);
}

// ctx[b,h] = sum_n resc[b,n] * word[b,n,h] ; write f16 into rnn_in/sel_in
__global__ void ctx_kernel(const float* __restrict__ resc, const float* __restrict__ word,
                           _Float16* __restrict__ rnn16, _Float16* __restrict__ sel16) {
  __shared__ float r[SW_];
  int b = blockIdx.x >> 1;
  int h = ((blockIdx.x & 1) << 8) + threadIdx.x;   // 256 threads, 2 blocks per batch row
  for (int i = threadIdx.x; i < SW_; i += 256) r[i] = resc[b * SW_ + i];
  __syncthreads();
  const float* wb = word + (size_t)b * SW_ * H_;
  float acc = 0.0f;
  for (int n = 0; n < SW_; ++n) acc += r[n] * wb[(size_t)n * H_ + h];
  _Float16 c = (_Float16)acc;
  rnn16[b * (2 * H_) + H_ + h]      = c;
  sel16[b * (3 * H_) + 2 * H_ + h]  = c;
}

// GRU gate: h_new from gi(16x1536), gh(16x1536), h(16x512); h updated in place
__global__ void gru_gate_kernel(const float* __restrict__ gi, const float* __restrict__ gh,
                                float* __restrict__ h32, _Float16* __restrict__ h16,
                                _Float16* __restrict__ sel16) {
  int b = blockIdx.x, hh = threadIdx.x;            // 512 threads
  const float* gib = gi + (size_t)b * 3 * H_;
  const float* ghb = gh + (size_t)b * 3 * H_;
  float ir = gib[hh], iz = gib[H_ + hh], in_ = gib[2 * H_ + hh];
  float hr = ghb[hh], hz = ghb[H_ + hh], hn  = ghb[2 * H_ + hh];
  float r = 1.0f / (1.0f + expf(-(ir + hr)));
  float z = 1.0f / (1.0f + expf(-(iz + hz)));
  float n = tanhf(in_ + r * hn);
  float hp = h32[b * H_ + hh];
  float hv = (1.0f - z) * n + z * hp;
  h32[b * H_ + hh] = hv;
  h16[b * H_ + hh] = (_Float16)hv;
  sel16[b * (3 * H_) + hh] = (_Float16)hv;
}

// p_gen[b] = sigmoid(sum_h tanh(selpre[b,h]) * vs_w[h] + vs_b)
__global__ void pgen_kernel(const float* __restrict__ selpre, const float* __restrict__ vsw,
                            const float* __restrict__ vsb, float* __restrict__ pgen,
                            float* __restrict__ out_pg, int t) {
  __shared__ float buf[32];
  int b = blockIdx.x, tid = threadIdx.x;           // 512 threads
  float v = tanhf(selpre[b * H_ + tid]) * vsw[tid];
  float s = block_sum(v, buf, 512);
  if (tid == 0) {
    float pg = 1.0f / (1.0f + expf(-(s + vsb[0])));
    pgen[b] = pg;
    out_pg[b * T_ + t] = pg;
  }
}

// row softmax over V, scale by p_gen, append extra_zeros -> out[b,t,:]
__global__ void vocab_kernel(const float* __restrict__ logits, const float* __restrict__ pgen,
                             const float* __restrict__ ez, float* __restrict__ out, int t) {
  __shared__ float buf[32];
  int b = blockIdx.x, tid = threadIdx.x;           // 1024 threads
  const float* row = logits + (size_t)b * V_;
  float mx = -3.4e38f;
  for (int i = tid; i < V_; i += 1024) mx = fmaxf(mx, row[i]);
  mx = block_max(mx, buf, 1024);
  float s = 0.0f;
  for (int i = tid; i < V_; i += 1024) s += expf(row[i] - mx);
  s = block_sum(s, buf, 1024);
  float scale = pgen[b] / s;
  float* dst = out + ((size_t)b * T_ + t) * VEXT_;
  for (int i = tid; i < V_; i += 1024) dst[i] = expf(row[i] - mx) * scale;
  for (int i = V_ + tid; i < VEXT_; i += 1024) dst[i] = ez[b * EXTRA_ + (i - V_)];
}

// scatter-add copy probabilities: out[b,t,safe_idx[b,n]] += (1-p_gen[b])*resc[b,n]
__global__ void scatter_kernel(const int* __restrict__ enc, const float* __restrict__ resc,
                               const float* __restrict__ pgen, float* __restrict__ out, int t) {
  int b = blockIdx.x, n = threadIdx.x;             // 512 threads
  int idx = enc[b * SW_ + n];
  if (idx >= VEXT_) idx = 0;
  float val = (1.0f - pgen[b]) * resc[b * SW_ + n];
  atomicAdd(out + ((size_t)b * T_ + t) * VEXT_ + idx, val);
}

__global__ void copy_kernel(const float* __restrict__ src, float* __restrict__ dst, int n) {
  int i = blockIdx.x * blockDim.x + threadIdx.x;
  if (i < n) dst[i] = src[i];
}

// ---------------- host launch ----------------
extern "C" void kernel_launch(void* const* d_in, const int* in_sizes, int n_in,
                              void* d_out, int out_size, void* d_ws, size_t ws_size,
                              hipStream_t stream) {
  const float* word = (const float*)d_in[0];
  const float* sent = (const float*)d_in[1];
  const float* dec0 = (const float*)d_in[2];
  const float* ez   = (const float*)d_in[3];
  const int*   tgt  = (const int*)d_in[4];
  const int*   enc  = (const int*)d_in[5];
  const float* emb  = (const float*)d_in[6];
  const float* gih  = (const float*)d_in[7];
  const float* ghh  = (const float*)d_in[8];
  const float* bih  = (const float*)d_in[9];
  const float* bhh  = (const float*)d_in[10];
  const float* outw = (const float*)d_in[11];
  const float* outb = (const float*)d_in[12];
  const float* waw1 = (const float*)d_in[13];
  const float* wab1 = (const float*)d_in[14];
  const float* waw2 = (const float*)d_in[15];
  const float* wab2 = (const float*)d_in[16];
  const float* wav  = (const float*)d_in[17];
  const float* wavb = (const float*)d_in[18];
  const float* saw1 = (const float*)d_in[19];
  const float* sab1 = (const float*)d_in[20];
  const float* saw2 = (const float*)d_in[21];
  const float* sab2 = (const float*)d_in[22];
  const float* sav  = (const float*)d_in[23];
  const float* savb = (const float*)d_in[24];
  const float* selw = (const float*)d_in[25];
  const float* selb = (const float*)d_in[26];
  const float* vsw  = (const float*)d_in[27];
  const float* vsb  = (const float*)d_in[28];

  char* wsb = (char*)d_ws;
  _Float16* OUTW16 = (_Float16*)(wsb + OFF_OUTW16);
  _Float16* GIH16  = (_Float16*)(wsb + OFF_GIH16);
  _Float16* GHH16  = (_Float16*)(wsb + OFF_GHH16);
  _Float16* WAW1f  = (_Float16*)(wsb + OFF_WAW1);
  _Float16* WAW2f  = (_Float16*)(wsb + OFF_WAW2);
  _Float16* SAW1f  = (_Float16*)(wsb + OFF_SAW1);
  _Float16* SAW2f  = (_Float16*)(wsb + OFF_SAW2);
  _Float16* SELW16 = (_Float16*)(wsb + OFF_SELW16);
  _Float16* WORD16 = (_Float16*)(wsb + OFF_WORD16);
  _Float16* SENT16 = (_Float16*)(wsb + OFF_SENT16);
  float*    WAPRE  = (float*)(wsb + OFF_WAPRE);
  float*    SAPRE  = (float*)(wsb + OFF_SAPRE);
  float*    H32    = (float*)(wsb + OFF_H32);
  _Float16* H16    = (_Float16*)(wsb + OFF_H16);
  _Float16* RNN16  = (_Float16*)(wsb + OFF_RNN16);
  _Float16* SEL16  = (_Float16*)(wsb + OFF_SEL16);
  float*    DQW    = (float*)(wsb + OFF_DQW);
  float*    DQS    = (float*)(wsb + OFF_DQS);
  float*    SCW    = (float*)(wsb + OFF_SCW);
  float*    SCS    = (float*)(wsb + OFF_SCS);
  float*    RESC   = (float*)(wsb + OFF_RESC);
  float*    GI     = (float*)(wsb + OFF_GI);
  float*    GH     = (float*)(wsb + OFF_GH);
  float*    SELPRE = (float*)(wsb + OFF_SELPRE);
  float*    PGEN   = (float*)(wsb + OFF_PGEN);
  float*    LOGITS = (float*)(wsb + OFF_LOGITS);

  float* out = (float*)d_out;

  auto cvt = [&](const float* src, _Float16* dst, size_t n) {
    int blocks = (int)((n + 255) / 256);
    if (blocks > 4096) blocks = 4096;
    f32_to_f16_kernel<<<blocks, 256, 0, stream>>>(src, dst, n);
  };
  auto gemm16 = [&](const _Float16* A, const _Float16* Wt, const float* bias,
                    float* o, int N, int K) {
    int blocks = ((N >> 4) * 32 + 255) / 256;
    gemm16_f16_kernel<<<blocks, 256, 0, stream>>>(A, Wt, bias, o, N, K);
  };

  // ---- one-time precompute ----
  cvt(outw, OUTW16, (size_t)V_ * H_);
  cvt(gih,  GIH16,  (size_t)3 * H_ * 2 * H_);
  cvt(ghh,  GHH16,  (size_t)3 * H_ * H_);
  cvt(waw1, WAW1f,  (size_t)H_ * H_);
  cvt(waw2, WAW2f,  (size_t)H_ * H_);
  cvt(saw1, SAW1f,  (size_t)H_ * H_);
  cvt(saw2, SAW2f,  (size_t)H_ * H_);
  cvt(selw, SELW16, (size_t)H_ * 3 * H_);
  cvt(word, WORD16, (size_t)NWROW_ * H_);
  cvt(sent, SENT16, (size_t)B_ * S_ * H_);

  {  // wa_pre = word(8192x512) @ wa_w1^T + b ; sa_pre = sent(256x512) @ sa_w1^T + b
    int tiles = (NWROW_ >> 4) * (H_ >> 4);
    gemm_tiled_f16_kernel<<<(tiles * 32 + 255) / 256, 256, 0, stream>>>(
        WORD16, WAW1f, wab1, WAPRE, NWROW_, H_, H_);
    tiles = ((B_ * S_) >> 4) * (H_ >> 4);
    gemm_tiled_f16_kernel<<<(tiles * 32 + 255) / 256, 256, 0, stream>>>(
        SENT16, SAW1f, sab1, SAPRE, B_ * S_, H_, H_);
  }
  init_h_kernel<<<B_, H_, 0, stream>>>(dec0, H32, H16);

  // ---- sequential decode ----
  for (int t = 0; t < T_; ++t) {
    embed_kernel<<<B_, H_, 0, stream>>>(tgt, emb, RNN16, SEL16, t);

    gemm16(H16, WAW2f, wab2, DQW, H_, H_);
    gemm16(H16, SAW2f, sab2, DQS, H_, H_);

    attn_score_kernel<<<(NWROW_ * 32 + 255) / 256, 256, 0, stream>>>(
        WAPRE, DQW, wav, wavb, SCW, SW_, NWROW_);
    attn_score_kernel<<<(B_ * S_ * 32 + 255) / 256, 256, 0, stream>>>(
        SAPRE, DQS, sav, savb, SCS, S_, B_ * S_);

    attn_combine_kernel<<<B_, SW_, 0, stream>>>(SCW, SCS, RESC);
    ctx_kernel<<<B_ * 2, 256, 0, stream>>>(RESC, word, RNN16, SEL16);

    gemm16(RNN16, GIH16, bih, GI, 3 * H_, 2 * H_);
    gemm16(H16,   GHH16, bhh, GH, 3 * H_, H_);
    gru_gate_kernel<<<B_, H_, 0, stream>>>(GI, GH, H32, H16, SEL16);

    gemm16(SEL16, SELW16, selb, SELPRE, H_, 3 * H_);
    pgen_kernel<<<B_, H_, 0, stream>>>(SELPRE, vsw, vsb, PGEN, out + OUT_OFF_PG, t);

    gemm16(H16, OUTW16, outb, LOGITS, V_, H_);   // dominant GEMM: 16x32000x512
    vocab_kernel<<<B_, 1024, 0, stream>>>(LOGITS, PGEN, ez, out, t);
    scatter_kernel<<<B_, SW_, 0, stream>>>(enc, RESC, PGEN, out, t);
  }

  copy_kernel<<<(B_ * H_ + 255) / 256, 256, 0, stream>>>(H32, out + OUT_OFF_H, B_ * H_);
}
